// SequentialLoRALinear_82437602279898
// MI455X (gfx1250) — matrix operand order repaired and checked by reading, not compiled, not run on hardware
//
#include <hip/hip_runtime.h>

// Problem constants (from reference)
#define T_DIM   8192
#define D_IN    4096
#define D_OUT   4096
#define R_LORA  16
#define NLORA   128     // K_LORA * R_LORA

typedef __attribute__((ext_vector_type(16))) __bf16 v16bf;
typedef __attribute__((ext_vector_type(2)))  __bf16 v2bf;
typedef __attribute__((ext_vector_type(8)))  float  v8f;

#if defined(__gfx1250__) && __has_builtin(__builtin_amdgcn_global_load_async_to_lds_b64) && \
    __has_builtin(__builtin_amdgcn_s_wait_asynccnt)
#define USE_ASYNC_LDS 1
typedef int v2i_async __attribute__((ext_vector_type(2)));
typedef __attribute__((address_space(1))) v2i_async* async_gptr;
typedef __attribute__((address_space(3))) v2i_async* async_lptr;
#else
#define USE_ASYNC_LDS 0
#endif

union Frag {            // 16 bf16 = 32 bytes = 8 VGPRs, assembled from two 16B LDS loads
    uint4 q[2];
    v16bf v;
};

// native f32 -> bf16 converts (let the compiler pick v_cvt_pk_bf16_f32 etc.)
__device__ __forceinline__ unsigned short f2bf(float f) {
    __bf16 h = (__bf16)f;
    return __builtin_bit_cast(unsigned short, h);
}
__device__ __forceinline__ unsigned pack2(float lo, float hi) {
    v2bf v = { (__bf16)lo, (__bf16)hi };
    return __builtin_bit_cast(unsigned, v);
}

#define LDA 40   // row stride (bf16 units) for 32-wide tiles; +8 pad -> conflict-free ds_load_b128

// One 128x128x32 MMA step: each wave issues 8 x v_wmma_f32_16x16x32_bf16.
__device__ __forceinline__ void mma_step(const unsigned short* Xs, const unsigned short* Ws,
                                         int wm, int wn, int lane, v8f acc[2][4]) {
    const int l16 = lane & 15;
    const int lh  = lane >> 4;
    Frag a[2], b[4];
#pragma unroll
    for (int ms = 0; ms < 2; ++ms) {
        // A 16x32: lane = row (lane&15); K chunks at (lane>>4)*8 and +16
        const unsigned short* p = &Xs[(wm * 32 + ms * 16 + l16) * LDA + lh * 8];
        a[ms].q[0] = *reinterpret_cast<const uint4*>(p);
        a[ms].q[1] = *reinterpret_cast<const uint4*>(p + 16);
    }
#pragma unroll
    for (int ns = 0; ns < 4; ++ns) {
        // B 32x16: lane = col (lane&15); K chunks at (lane>>4)*8 and +16
        const unsigned short* p = &Ws[(wn * 64 + ns * 16 + l16) * LDA + lh * 8];
        b[ns].q[0] = *reinterpret_cast<const uint4*>(p);
        b[ns].q[1] = *reinterpret_cast<const uint4*>(p + 16);
    }
#pragma unroll
    for (int ms = 0; ms < 2; ++ms)
#pragma unroll
        for (int ns = 0; ns < 4; ++ns)
            acc[ms][ns] = __builtin_amdgcn_wmma_f32_16x16x32_bf16(
                false, a[ms].v, false, b[ns].v, (short)0, acc[ms][ns], false, false);
}

// MODE 0: Bx = x @ B_all^T            (out = bf16 [T,128] in workspace)
// MODE 1: out = x@W^T + Bx@Aall^T + b (out = f32  [T,D_OUT])
template <int MODE>
__global__ __launch_bounds__(256) void lora_gemm_wmma(
    const float* __restrict__ X,            // [T, D_IN] f32
    const float* __restrict__ Wm,           // [Nrows, D_IN] f32 (W or flattened B_stack)
    const float* __restrict__ bias,         // [D_OUT]            (MODE 1)
    const float* __restrict__ Astk,         // [K, D_OUT, R]      (MODE 1)
    const unsigned short* __restrict__ Bx,  // bf16 [T, 128]      (MODE 1)
    void* __restrict__ outp) {
    __shared__ __align__(16) unsigned short Xs[2][128 * LDA];   // double-buffered
    __shared__ __align__(16) unsigned short Ws[2][128 * LDA];

    const int tid  = threadIdx.x;
    const int lane = tid & 31;
    const int wave = tid >> 5;
    const int wm   = wave >> 1;     // 0..3  (M sub-block of 32)
    const int wn   = wave & 1;      // 0..1  (N sub-block of 64)
    const int l16  = lane & 15;
    const int lh   = lane >> 4;
    const int t0   = blockIdx.x * 128;
    const int o0   = blockIdx.y * 128;
    const int rowb = tid >> 3;          // base row of this thread's loads
    const int col  = (tid & 7) << 2;    // f32/bf16 column within 32-wide k-tile

    v8f acc[2][4];
#pragma unroll
    for (int ms = 0; ms < 2; ++ms)
#pragma unroll
        for (int ns = 0; ns < 4; ++ns)
#pragma unroll
            for (int v = 0; v < 8; ++v) acc[ms][ns][v] = 0.0f;

    float4 xr[4], wr[4];

    auto load_global = [&](int kk) {
#pragma unroll
        for (int it = 0; it < 4; ++it) {
            int row = it * 32 + rowb;
            const float* xp = X  + (size_t)(t0 + row) * D_IN + kk + col;
            const float* wp = Wm + (size_t)(o0 + row) * D_IN + kk + col;
            xr[it] = *reinterpret_cast<const float4*>(xp);
            wr[it] = *reinterpret_cast<const float4*>(wp);
            if (kk + 64 < D_IN) {               // gfx1250 global_prefetch, 2 tiles ahead
                __builtin_prefetch(xp + 64, 0, 0);
                __builtin_prefetch(wp + 64, 0, 0);
            }
        }
    };
    auto store_lds = [&](int buf) {
#pragma unroll
        for (int it = 0; it < 4; ++it) {
            int row = it * 32 + rowb;
            *reinterpret_cast<uint2*>(&Xs[buf][row * LDA + col]) =
                make_uint2(pack2(xr[it].x, xr[it].y), pack2(xr[it].z, xr[it].w));
            *reinterpret_cast<uint2*>(&Ws[buf][row * LDA + col]) =
                make_uint2(pack2(wr[it].x, wr[it].y), pack2(wr[it].z, wr[it].w));
        }
    };

    // ---- main K loop over D_IN, double-buffered through LDS ----
    load_global(0);
    store_lds(0);
    const int nIter = D_IN / 32;
#pragma unroll 2
    for (int k = 0; k < nIter; ++k) {
        __syncthreads();
        if (k + 1 < nIter) load_global((k + 1) * 32);       // in flight during WMMAs
        mma_step(Xs[k & 1], Ws[k & 1], wm, wn, lane, acc);
        if (k + 1 < nIter) store_lds((k + 1) & 1);          // fill other buffer
    }

    if (MODE == 1) {
        // ---- LoRA epilogue: 128 extra K-dims. A-op = Bx (bf16), B-op = Aall^T gathered ----
        for (int j0 = 0; j0 < NLORA; j0 += 32) {
            __syncthreads();
#if USE_ASYNC_LDS
#pragma unroll
            for (int it = 0; it < 4; ++it) {    // Bx tile via async global->LDS (ASYNCcnt)
                int row = it * 32 + rowb;
                const unsigned short* gp = Bx + (size_t)(t0 + row) * NLORA + j0 + col;
                unsigned short* lp = &Xs[0][row * LDA + col];
                __builtin_amdgcn_global_load_async_to_lds_b64(
                    (async_gptr)gp, (async_lptr)lp, 0, 0);
            }
#else
#pragma unroll
            for (int it = 0; it < 4; ++it) {    // Bx tile: 128 rows x 32 (already bf16)
                int row = it * 32 + rowb;
                uint2 d = *reinterpret_cast<const uint2*>(Bx + (size_t)(t0 + row) * NLORA + j0 + col);
                *reinterpret_cast<uint2*>(&Xs[0][row * LDA + col]) = d;
            }
#endif
#pragma unroll
            for (int it = 0; it < 16; ++it) {   // Aall^T tile: gather A_stack[k, o0+n, r]
                int idx = it * 256 + tid;
                int n   = idx >> 5;
                int j   = j0 + (idx & 31);
                float av = Astk[(size_t)(j >> 4) * D_OUT * R_LORA +
                                (size_t)(o0 + n) * R_LORA + (j & 15)];
                Ws[0][n * LDA + (idx & 31)] = f2bf(av);
            }
#if USE_ASYNC_LDS
            __builtin_amdgcn_s_wait_asynccnt(0);
#endif
            __syncthreads();
            mma_step(Xs[0], Ws[0], wm, wn, lane, acc);
        }
        // ---- bias + store f32 ----
        float* out = reinterpret_cast<float*>(outp);
#pragma unroll
        for (int ms = 0; ms < 2; ++ms)
#pragma unroll
            for (int ns = 0; ns < 4; ++ns) {
                int n = o0 + wn * 64 + ns * 16 + l16;
                float bb = bias[n];
#pragma unroll
                for (int v = 0; v < 8; ++v) {
                    int row = t0 + wm * 32 + ms * 16 + v + 8 * lh;  // C layout: M = v + 8*(lane>>4)
                    out[(size_t)row * D_OUT + n] = acc[ms][ns][v] + bb;
                }
            }
    } else {
        // ---- store Bx as bf16 [T,128] ----
        unsigned short* bx = reinterpret_cast<unsigned short*>(outp);
#pragma unroll
        for (int ms = 0; ms < 2; ++ms)
#pragma unroll
            for (int ns = 0; ns < 4; ++ns) {
                int n = wn * 64 + ns * 16 + l16;
#pragma unroll
                for (int v = 0; v < 8; ++v) {
                    int row = t0 + wm * 32 + ms * 16 + v + 8 * lh;
                    bx[(size_t)row * NLORA + n] = f2bf(acc[ms][ns][v]);
                }
            }
    }
}

extern "C" void kernel_launch(void* const* d_in, const int* in_sizes, int n_in,
                              void* d_out, int out_size, void* d_ws, size_t ws_size,
                              hipStream_t stream) {
    (void)in_sizes; (void)n_in; (void)out_size; (void)ws_size;
    const float* x  = (const float*)d_in[0];   // [T, D_IN]
    const float* W  = (const float*)d_in[1];   // [D_OUT, D_IN]
    const float* b  = (const float*)d_in[2];   // [D_OUT]
    const float* As = (const float*)d_in[3];   // [K, D_OUT, R]
    const float* Bs = (const float*)d_in[4];   // [K, R, D_IN]  == flat [128, D_IN]
    float* out = (float*)d_out;
    unsigned short* Bx = (unsigned short*)d_ws;  // bf16 [T, 128] scratch (2 MB)

    // Stage 1: Bx = x @ B_all^T   (N = 128 = one block tile wide)
    lora_gemm_wmma<0><<<dim3(T_DIM / 128, 1), 256, 0, stream>>>(
        x, Bs, nullptr, nullptr, nullptr, (void*)Bx);
    // Stage 2: out = x @ W^T + Bx @ Aall^T + b, fully fused in WMMA accumulators
    lora_gemm_wmma<1><<<dim3(T_DIM / 128, D_OUT / 128), 256, 0, stream>>>(
        x, W, b, As, Bx, (void*)out);
}
